// TimeLSTM_45346264711222
// MI455X (gfx1250) — matrix operand order repaired and checked by reading, not compiled, not run on hardware
//
#include <hip/hip_runtime.h>

#define B_   256
#define S_   512
#define I_   64
#define HS_  128
#define NX_  640   // 5*HS
#define NH_  384   // 3*HS
#define ROWS 16    // batch rows per workgroup (WMMA M)
#define THREADS 256
#define WAVES 8

typedef __attribute__((ext_vector_type(16))) _Float16 v16h;
typedef __attribute__((ext_vector_type(8)))  _Float16 v8h;
typedef __attribute__((ext_vector_type(8)))  float    v8f;
typedef __attribute__((ext_vector_type(4)))  float    v4f;

// ---- LDS layout (bytes). Total 323,584 B <= 320 KB/WGP ----
#define WX_OFF 0        //  64x640 f16, B-frag swizzled: 81920 B
#define WH_OFF 81920    // 128x384 f16: 98304 B
#define WT_OFF 180224   //  64x384 f16 (clamped col<HS): 49152 B
#define GX_OFF 229376   // 16x640 f32: 40960 B
#define GT_OFF 270336   // 16x384 f32: 24576 B
#define XH_OFF 294912   // 16x384 f32: 24576 B
#define HB_OFF 319488   // 16x128 f16 h-state: 4096 B
#define LDS_BYTES 323584

__device__ __forceinline__ float sigmoidf_(float x) {
  return 1.0f / (1.0f + __expf(-x));
}
__device__ __forceinline__ float tanhf_(float x) {
  return 2.0f / (1.0f + __expf(-2.0f * x)) - 1.0f;
}

// Convert f32 weights (row-major KxN) into f16 LDS, pre-swizzled into the
// CDNA5 WMMA f16 B-matrix (32x16 per tile) lane layout:
//   lane l holds N = l%16; K = (l<16 ? 0 : 8) + (e<8 ? e : 8 + e)  (e = half-element 0..15)
// Each (ntile, ktile) block is 32 lanes * 16 halves = 1024 B, so a B-fragment
// load is one contiguous 32-byte LDS read per lane.
__device__ __forceinline__ void stage_weights(const float* __restrict__ W,
                                              _Float16* dst, int K, int N,
                                              int tid, bool clamp_first) {
  const int nkt = K >> 5;
  const int total = K * N;
  for (int idx = tid; idx < total; idx += THREADS) {
    int k = idx / N;
    int n = idx - k * N;
    float v = W[idx];
    if (clamp_first && n < HS_) v = fminf(v, 0.0f);   // wt[:, :hs] = min(wt, 0)
    int kt = k >> 5, kk = k & 31;
    int ntile = n >> 4, nn = n & 15;
    int l = ((kk >> 3) & 1) ? (16 + nn) : nn;
    int e = (kk & 7) | ((kk & 16) >> 1);
    dst[((size_t)(ntile * nkt + kt) * 32 + l) * 16 + e] = (_Float16)v;
  }
}

// Build a 16x32 f16 A-fragment from global f32 (row stride S_*I_ between
// batch rows). ISA A layout: lane l holds M = l%16, K runs
// [base, base+7] and [base+16, base+23], base = (l<16 ? 0 : 8).
__device__ __forceinline__ v16h a_from_global(const float* __restrict__ p0, int lane) {
  const int m = lane & 15;
  const int kb = (lane < 16) ? 0 : 8;
  const float* p = p0 + (size_t)m * (S_ * I_) + kb;
  v4f x0 = *(const v4f*)(p);
  v4f x1 = *(const v4f*)(p + 4);
  v4f x2 = *(const v4f*)(p + 16);
  v4f x3 = *(const v4f*)(p + 20);
  v16h a;
#pragma unroll
  for (int i = 0; i < 4; ++i) {
    a[i]      = (_Float16)x0[i];
    a[4 + i]  = (_Float16)x1[i];
    a[8 + i]  = (_Float16)x2[i];
    a[12 + i] = (_Float16)x3[i];
  }
  return a;
}

// Build a 16x32 A-fragment from the LDS f16 h-state (row-major 16x128).
__device__ __forceinline__ v16h a_from_lds_h(const _Float16* hb, int ktile, int lane) {
  const int m = lane & 15;
  const int kb = ktile * 32 + ((lane < 16) ? 0 : 8);
  const _Float16* p = hb + m * HS_ + kb;
  v8h lo = *(const v8h*)(p);
  v8h hi = *(const v8h*)(p + 16);
  v16h a;
#pragma unroll
  for (int i = 0; i < 8; ++i) { a[i] = lo[i]; a[8 + i] = hi[i]; }
  return a;
}

// One 16x16 output tile. All NKT B-fragments are loaded up-front (independent
// ds_load_b128 pairs), then the WMMA accumulation chain runs, then the f32
// D-fragment (lane l: N=l%16, M=(l<16?0:8)+r) is scattered to LDS. Fully
// inlined + unrolled at the call sites so the scheduler can overlap the next
// tile's LDS loads / previous tile's stores with the WMMA chain instead of
// padding hazards with v_nop.
template <int NKT>
__device__ __forceinline__ void gemm_tile(const _Float16* wseg, int blk0,
                                          const v16h* afr, float* dst, int ldn,
                                          int ntile, int lane) {
  v16h bf[NKT];
#pragma unroll
  for (int kt = 0; kt < NKT; ++kt)
    bf[kt] = *(const v16h*)(wseg + (size_t)(blk0 + kt) * 512 + lane * 16);
  v8f acc = {};
#pragma unroll
  for (int kt = 0; kt < NKT; ++kt)
    acc = __builtin_amdgcn_wmma_f32_16x16x32_f16(
        /*neg_a=*/false, afr[kt], /*neg_b=*/false, bf[kt],
        /*c_mod=*/(short)0, acc, /*reuse_a=*/false, /*reuse_b=*/false);
  const int n0 = ntile * 16 + (lane & 15);
  const int rb = (lane < 16) ? 0 : 8;
#pragma unroll
  for (int r = 0; r < 8; ++r) dst[(size_t)(rb + r) * ldn + n0] = acc[r];
}

__global__ __launch_bounds__(THREADS)
void time_lstm_persistent(const float* __restrict__ x, const float* __restrict__ td,
                          const float* __restrict__ Wx, const float* __restrict__ Wh,
                          const float* __restrict__ Wt, const float* __restrict__ bias,
                          float* __restrict__ out) {
  extern __shared__ char smem[];
  _Float16* wx  = (_Float16*)(smem + WX_OFF);
  _Float16* wh  = (_Float16*)(smem + WH_OFF);
  _Float16* wt  = (_Float16*)(smem + WT_OFF);
  float*    gxs = (float*)(smem + GX_OFF);
  float*    gts = (float*)(smem + GT_OFF);
  float*    xhs = (float*)(smem + XH_OFF);
  _Float16* hb  = (_Float16*)(smem + HB_OFF);

  const int tid  = threadIdx.x;
  const int lane = tid & 31;
  // Force the wave index into an SGPR: the per-wave tile bases below become
  // scalar, so the GEMM phase has zero vector-predicated control flow and
  // EXEC is trivially all-ones around every WMMA (ISA requirement).
  const int wave = __builtin_amdgcn_readfirstlane(tid >> 5);
  const int b0   = blockIdx.x * ROWS;

  // One-time: stage all weights as f16, pre-swizzled to B-fragment layout.
  stage_weights(Wx, wx, I_,  NX_, tid, false);
  stage_weights(Wh, wh, HS_, NH_, tid, false);
  stage_weights(Wt, wt, I_,  NH_, tid, true);

  for (int i = tid; i < ROWS * HS_; i += THREADS) hb[i] = (_Float16)0.0f;

  // Per-thread gate lanes: column col, rows rbase..rbase+7.
  const int col   = tid & (HS_ - 1);
  const int rbase = (tid >> 7) * 8;
  float creg[8], hreg[8];
#pragma unroll
  for (int r = 0; r < 8; ++r) { creg[r] = 0.0f; hreg[r] = 0.0f; }
  const float bb0 = bias[col];
  const float bb1 = bias[HS_ + col];
  const float bb2 = bias[2 * HS_ + col];
  const float bb3 = bias[3 * HS_ + col];
  const float bb4 = bias[4 * HS_ + col];

  // Static per-wave tile slices: 8 waves x (5 gx + 3 gt + 3 xh) = 88 tiles.
  const int gx_nt0 = wave * 5;   // gx: 40 N-tiles of 640
  const int gt_nt0 = wave * 3;   // gt: 24 N-tiles of 384
  const int xh_nt0 = wave * 3;   // xh: 24 N-tiles of 384

  __syncthreads();

  for (int t = 0; t < S_; ++t) {
    const float* xb = x  + ((size_t)b0 * S_ + t) * I_;
    const float* tb = td + ((size_t)b0 * S_ + t) * I_;
    __builtin_prefetch(xb + I_, 0, 0);   // global_prefetch next timestep slice
    __builtin_prefetch(tb + I_, 0, 0);

    // A fragments: x/TimeDiff from global, h from LDS (prev step's output).
    v16h ax[2], at[2], ah[4];
#pragma unroll
    for (int kt = 0; kt < 2; ++kt) {
      ax[kt] = a_from_global(xb + kt * 32, lane);
      at[kt] = a_from_global(tb + kt * 32, lane);
    }
#pragma unroll
    for (int kt = 0; kt < 4; ++kt) ah[kt] = a_from_lds_h(hb, kt, lane);

    // Branch-free GEMM phase: fully unrolled per-wave tile lists.
#pragma unroll
    for (int i = 0; i < 5; ++i) {
      const int nt = gx_nt0 + i;
      gemm_tile<2>(wx, nt * 2, ax, gxs, NX_, nt, lane);
    }
#pragma unroll
    for (int i = 0; i < 3; ++i) {
      const int nt = gt_nt0 + i;
      gemm_tile<2>(wt, nt * 2, at, gts, NH_, nt, lane);
    }
#pragma unroll
    for (int i = 0; i < 3; ++i) {
      const int nt = xh_nt0 + i;
      gemm_tile<4>(wh, nt * 4, ah, xhs, NH_, nt, lane);
    }
    __syncthreads();

    // Gate math (f32, transcendentals) + h/c update.
#pragma unroll
    for (int rr = 0; rr < 8; ++rr) {
      const int r = rbase + rr;
      const float gx0 = gxs[r * NX_ + col];
      const float gx1 = gxs[r * NX_ + HS_ + col];
      const float gx2 = gxs[r * NX_ + 2 * HS_ + col];
      const float gx3 = gxs[r * NX_ + 3 * HS_ + col];
      const float gx4 = gxs[r * NX_ + 4 * HS_ + col];
      const float gt0 = gts[r * NH_ + col];
      const float gt1 = gts[r * NH_ + HS_ + col];
      const float gt2 = gts[r * NH_ + 2 * HS_ + col];
      const float xh0 = xhs[r * NH_ + col];
      const float xh1 = xhs[r * NH_ + HS_ + col];
      const float xh2 = xhs[r * NH_ + 2 * HS_ + col];
      const float c   = creg[rr];

      const float i_t  = sigmoidf_(gx0 + xh0 + bb0);
      const float t1   = sigmoidf_(gx1 + tanhf_(gt0) + bb1);
      const float t2   = sigmoidf_(gx2 + tanhf_(gt1) + bb2);
      const float cwa  = tanhf_(gx3 + xh1 + bb3);
      const float it1  = i_t * t1;
      const float ctil = sigmoidf_((1.0f - it1) * c + it1 * cwa);
      const float cnew = sigmoidf_((1.0f - i_t) * c + i_t * t2 * cwa);
      const float o_t  = sigmoidf_(gx4 + gt2 + xh2 + bb4);
      const float hnew = o_t + tanhf_(ctil);

      creg[rr] = cnew;
      hreg[rr] = hnew;
      hb[r * HS_ + col] = (_Float16)hnew;                       // next-step A
      out[(((size_t)(b0 + r) * S_) + t) * HS_ + col] = hnew;    // hidden_seq
    }
    __syncthreads();
  }

  // Finals: (h_f, c_f) appended after hidden_seq.
  float* hf = out + (size_t)B_ * S_ * HS_;
  float* cf = hf + (size_t)B_ * HS_;
#pragma unroll
  for (int rr = 0; rr < 8; ++rr) {
    const int r = rbase + rr;
    hf[(size_t)(b0 + r) * HS_ + col] = hreg[rr];
    cf[(size_t)(b0 + r) * HS_ + col] = creg[rr];
  }
}

extern "C" void kernel_launch(void* const* d_in, const int* in_sizes, int n_in,
                              void* d_out, int out_size, void* d_ws, size_t ws_size,
                              hipStream_t stream) {
  (void)in_sizes; (void)n_in; (void)out_size; (void)d_ws; (void)ws_size;
  const float* x    = (const float*)d_in[0];
  const float* td   = (const float*)d_in[1];
  const float* Wx   = (const float*)d_in[2];
  const float* Wh   = (const float*)d_in[3];
  const float* Wt   = (const float*)d_in[4];
  const float* bias = (const float*)d_in[5];
  float* out = (float*)d_out;

  dim3 grid(B_ / ROWS);   // 16 persistent workgroups, one per 16-row batch tile
  dim3 block(THREADS);    // 8 waves (wave32)
  time_lstm_persistent<<<grid, block, LDS_BYTES, stream>>>(x, td, Wx, Wh, Wt, bias, out);
}
